// CrossDomainAttention_68831145886577
// MI455X (gfx1250) — compile-verified, matrix operand order checked by reading, not run on hardware
//
#include <hip/hip_runtime.h>
#include <hip/hip_bf16.h>
#include <math.h>

#define BATCH 16
#define LDIM  512
#define CDIM  2048

typedef __attribute__((ext_vector_type(16))) __bf16       v16bf;
typedef __attribute__((ext_vector_type(8)))  float        v8f;
typedef __attribute__((ext_vector_type(4)))  unsigned int u32x4;

union FragBF { v16bf bf; u32x4 q[2]; };
union U4H8   { u32x4 v; unsigned short h[8]; };

__device__ __forceinline__ unsigned short f2bf(float f) {
  unsigned int u = __builtin_bit_cast(unsigned int, f);
  u += 0x7FFFu + ((u >> 16) & 1u);   // round-to-nearest-even
  return (unsigned short)(u >> 16);
}

// --------------------------- TDM support -----------------------------------
#if defined(__AMDGCN__) && __has_builtin(__builtin_amdgcn_tensor_load_to_lds) && \
    __has_builtin(__builtin_amdgcn_s_wait_tensorcnt)
#define USE_TDM 1
#else
#define USE_TDM 0
#endif

#if USE_TDM
typedef __attribute__((ext_vector_type(8))) int v8i;
typedef __attribute__((ext_vector_type(4))) int v4i;

// 2D tile load: tileRows x 512 bf16 elements, row stride 512 elements in
// memory, LDS rows padded to 520 halfwords via TDM pad (16B after each 1024B
// row).  Issued by one wave; completion tracked on TENSORcnt.
__device__ __forceinline__ void tdm_load_tile_2d(unsigned ldsAddr,
                                                 const unsigned short* gptr,
                                                 int tileRows) {
  const unsigned long long ga = (unsigned long long)gptr;
  u32x4 g0;
  g0.x = 1u;                                                  // count=1
  g0.y = ldsAddr;                                             // lds_addr bytes
  g0.z = (unsigned)ga;                                        // global_addr[31:0]
  g0.w = (unsigned)((ga >> 32) & 0x01FFFFFFull) | (2u << 30); // addr[56:32]|type=2
  v8i g1;
  g1[0] = (int)((1u << 16)     // data_size = 2 bytes
              | (1u << 20)     // pad_enable
              | (7u << 22)     // pad_interval = 256 DWORDs (one row)
              | (3u << 25));   // pad_amount   = 4 DWORDs (16 B)
  g1[1] = (int)(512u << 16);   // tensor_dim0 = 512 elements  (bits 79:48 lo)
  g1[2] = (int)(2048u << 16);  // tensor_dim1 = 2048 rows     (bits 111:80 lo)
  g1[3] = (int)(512u << 16);   // tile_dim0 = 512             (bits 127:112)
  g1[4] = tileRows;            // tile_dim1                   (bits 143:128)
  g1[5] = 512;                 // tensor_dim0_stride = 512 elements
  g1[6] = 0;
  g1[7] = 0;
  v4i gz = {0, 0, 0, 0};
#if __clang_major__ >= 23
  v8i gz8 = {0, 0, 0, 0, 0, 0, 0, 0};
  __builtin_amdgcn_tensor_load_to_lds(g0, g1, gz, gz, gz8, 0);
#else
  __builtin_amdgcn_tensor_load_to_lds(g0, g1, gz, gz, 0);
#endif
}
#endif  // USE_TDM

// ---------------------------------------------------------------------------
// Kernel A: fused QKV projection.  X = transpose(a or b)  (C x L per batch)
//   out = X @ W^T + bias   (Q additionally scaled by 1/sqrt(L)), stored bf16.
// grid: (C/64, L/64, 3*B), block 128 (4 waves). Wave w owns rows [16w,16w+16).
// ---------------------------------------------------------------------------
__global__ __launch_bounds__(128)
void qkv_proj_kernel(const float* __restrict__ a, const float* __restrict__ bsrc,
                     const float* __restrict__ Wq, const float* __restrict__ bq,
                     const float* __restrict__ Wk, const float* __restrict__ bk,
                     const float* __restrict__ Wv, const float* __restrict__ bv,
                     unsigned short* __restrict__ wsQ,
                     unsigned short* __restrict__ wsK,
                     unsigned short* __restrict__ wsV)
{
  constexpr int LDT = 40;                       // 32 K + 8 pad (conflict-free b128)
  __shared__ unsigned short sA[64 * LDT];
  __shared__ unsigned short sB[64 * LDT];

  const int which = blockIdx.z / BATCH;         // 0=Q, 1=K, 2=V
  const int batch = blockIdx.z % BATCH;
  const int c0 = blockIdx.x * 64;               // output row block (c axis)
  const int n0 = blockIdx.y * 64;               // output feature block (m axis)

  const float* in   = (which == 0) ? a  : bsrc;
  const float* W    = (which == 0) ? Wq : (which == 1) ? Wk : Wv;
  const float* bias = (which == 0) ? bq : (which == 1) ? bk : bv;
  unsigned short* outp = (which == 0) ? wsQ : (which == 1) ? wsK : wsV;
  const float scale = (which == 0) ? 0.04419417382415922f : 1.0f;  // 1/sqrt(512)

  const int w    = threadIdx.x >> 5;
  const int lane = threadIdx.x & 31;
  const int r    = lane & 15;
  const int hi   = lane >> 4;
  const int hsel = hi * 8;

  v8f acc[4] = {};

  const size_t inBase = (size_t)batch * LDIM * CDIM;
  for (int k0 = 0; k0 < LDIM; k0 += 32) {
    // stage A tile: rows c (64) x k=l (32), transposed read of input, -> bf16
#pragma unroll 4
    for (int i = 0; i < 16; ++i) {
      int idx = threadIdx.x + i * 128;
      int ll = idx >> 6;                        // 0..31
      int cc = idx & 63;                        // 0..63 (coalesced)
      sA[cc * LDT + ll] = f2bf(in[inBase + (size_t)(k0 + ll) * CDIM + c0 + cc]);
    }
    // stage B tile: weight rows m (64) x l (32) -> bf16 (col-per-output-feature)
#pragma unroll 4
    for (int i = 0; i < 16; ++i) {
      int idx = threadIdx.x + i * 128;
      int mm = idx >> 5;                        // 0..63
      int ll = idx & 31;                        // coalesced within row
      sB[mm * LDT + ll] = f2bf(W[(size_t)(n0 + mm) * LDIM + k0 + ll]);
    }
    __syncthreads();

    FragBF fa;
    const unsigned short* pa = &sA[(w * 16 + r) * LDT + hsel];
    fa.q[0] = *(const u32x4*)pa;
    fa.q[1] = *(const u32x4*)(pa + 16);
#pragma unroll
    for (int nt = 0; nt < 4; ++nt) {
      FragBF fb;
      const unsigned short* pb = &sB[(nt * 16 + r) * LDT + hsel];
      fb.q[0] = *(const u32x4*)pb;
      fb.q[1] = *(const u32x4*)(pb + 16);
      acc[nt] = __builtin_amdgcn_wmma_f32_16x16x32_bf16(
          false, fa.bf, false, fb.bf, (short)0, acc[nt], false, false);
    }
    __syncthreads();
  }

  // epilogue: bias, scale, convert, store bf16 (layout: (b, c, m) row-major)
#pragma unroll
  for (int nt = 0; nt < 4; ++nt) {
    const int n = n0 + nt * 16 + r;
    const float bval = bias[n];
#pragma unroll
    for (int i = 0; i < 8; ++i) {
      const int crow = c0 + w * 16 + i + hi * 8;
      const float vv = (acc[nt][i] + bval) * scale;
      outp[((size_t)batch * CDIM + crow) * LDIM + n] = f2bf(vv);
    }
  }
}

// ---------------------------------------------------------------------------
// Kernel B: flash attention.  O = softmax(Q K^T) V  (scale folded into Q).
// grid: (C/64, B), block 256 (8 waves).  Wave (mg,ng): 16 q-rows x 256 feats.
// Q and K tiles staged by the Tensor Data Mover when available.
// ---------------------------------------------------------------------------
__global__ __launch_bounds__(256)
void attn_kernel(const unsigned short* __restrict__ wsQ,
                 const unsigned short* __restrict__ wsK,
                 const unsigned short* __restrict__ wsV,
                 float* __restrict__ out)
{
  constexpr int LQ = 520;                       // 512 + 8 pad
  constexpr int LP = 40;                        // 32 + 8 pad
  extern __shared__ unsigned short smem[];
  unsigned short* sQ  = smem;                   // 64 x 520
  unsigned short* sK  = sQ  + 64 * LQ;          // 32 x 520  (key-major: B-ready)
  unsigned short* sVt = sK  + 32 * LQ;          // 512 x 40  (feature-major)
  unsigned short* sP  = sVt + 512 * LP;         // 8 waves x 16 x 40

  const int b  = blockIdx.y;
  const int q0 = blockIdx.x * 64;
  const int w    = threadIdx.x >> 5;
  const int lane = threadIdx.x & 31;
  const int r    = lane & 15;
  const int hi   = lane >> 4;
  const int hsel = hi * 8;
  const int mg   = w >> 1;                      // 0..3: q-row group
  const int ng   = w & 1;                       // 0..1: feature half
  const int feat0 = ng * 256;

#if USE_TDM
  const unsigned ldsBase = __builtin_amdgcn_groupstaticsize();
  // stage Q block (64 rows x 512 bf16) once via TDM; waited with first K tile
  if (w == 0)
    tdm_load_tile_2d(ldsBase + (unsigned)((const char*)sQ - (const char*)smem),
                     wsQ + ((size_t)b * CDIM + q0) * LDIM, 64);
#else
  {
    const int row  = threadIdx.x >> 2;
    const int part = (threadIdx.x & 3) * 128;
    const unsigned short* g = wsQ + ((size_t)b * CDIM + q0 + row) * LDIM + part;
    unsigned short* d = sQ + row * LQ + part;
#pragma unroll
    for (int i = 0; i < 16; ++i)
      *(u32x4*)(d + i * 8) = *(const u32x4*)(g + i * 8);
  }
#endif

  v8f o[16] = {};
  float rowM[8], rowL[8];
#pragma unroll
  for (int i = 0; i < 8; ++i) { rowM[i] = -3.0e38f; rowL[i] = 0.f; }

  for (int kc = 0; kc < 64; ++kc) {
    __syncthreads();                            // prior readers done
#if USE_TDM
    if (w == 0)                                 // issue early: overlap with V stage
      tdm_load_tile_2d(ldsBase + (unsigned)((const char*)sK - (const char*)smem),
                       wsK + ((size_t)b * CDIM + kc * 32) * LDIM, 32);
#endif
    {
      const int key  = threadIdx.x >> 3;        // 0..31
      const int part = (threadIdx.x & 7) * 64;
#if !USE_TDM
      const unsigned short* gk = wsK + ((size_t)b * CDIM + kc * 32 + key) * LDIM + part;
      unsigned short* dk = sK + key * LQ + part;
#pragma unroll
      for (int i = 0; i < 8; ++i)
        *(u32x4*)(dk + i * 8) = *(const u32x4*)(gk + i * 8);
#endif
      const unsigned short* gv = wsV + ((size_t)b * CDIM + kc * 32 + key) * LDIM + part;
#pragma unroll
      for (int i = 0; i < 8; ++i) {             // transpose V into feature-major
        U4H8 t; t.v = *(const u32x4*)(gv + i * 8);
#pragma unroll
        for (int j = 0; j < 8; ++j)
          sVt[(part + i * 8 + j) * LP + key] = t.h[j];
      }
      if (kc + 1 < 64)
        __builtin_prefetch(wsV + ((size_t)b * CDIM + (kc + 1) * 32 + key) * LDIM + part, 0, 1);
    }
#if USE_TDM
    if (w == 0)
      __builtin_amdgcn_s_wait_tensorcnt(0);     // K (and initial Q) landed in LDS
#endif
    __syncthreads();

    // scores: S(16q x 32key) over 512-dim features
    v8f s[2] = {};
    const unsigned short* pq = sQ + (mg * 16 + r) * LQ + hsel;
#pragma unroll
    for (int kt = 0; kt < 2; ++kt) {
      const unsigned short* pk = sK + (kt * 16 + r) * LQ + hsel;
#pragma unroll
      for (int c2 = 0; c2 < LDIM; c2 += 32) {
        FragBF fa, fb;
        fa.q[0] = *(const u32x4*)(pq + c2);
        fa.q[1] = *(const u32x4*)(pq + c2 + 16);
        fb.q[0] = *(const u32x4*)(pk + c2);
        fb.q[1] = *(const u32x4*)(pk + c2 + 16);
        s[kt] = __builtin_amdgcn_wmma_f32_16x16x32_bf16(
            false, fa.bf, false, fb.bf, (short)0, s[kt], false, false);
      }
    }

    // online softmax: rows live per-vgpr, spread over 16-lane groups
    float alpha[8];
#pragma unroll
    for (int i = 0; i < 8; ++i) {
      float mx = fmaxf(s[0][i], s[1][i]);
      mx = fmaxf(mx, __shfl_xor(mx, 1, 16));
      mx = fmaxf(mx, __shfl_xor(mx, 2, 16));
      mx = fmaxf(mx, __shfl_xor(mx, 4, 16));
      mx = fmaxf(mx, __shfl_xor(mx, 8, 16));
      const float nm = fmaxf(rowM[i], mx);
      const float al = __expf(rowM[i] - nm);
      const float p0 = __expf(s[0][i] - nm);
      const float p1 = __expf(s[1][i] - nm);
      float sum = p0 + p1;
      sum += __shfl_xor(sum, 1, 16);
      sum += __shfl_xor(sum, 2, 16);
      sum += __shfl_xor(sum, 4, 16);
      sum += __shfl_xor(sum, 8, 16);
      rowL[i] = rowL[i] * al + sum;
      rowM[i] = nm;
      alpha[i] = al;
      s[0][i] = p0;
      s[1][i] = p1;
    }

    // D-layout -> A-layout via per-wave LDS patch (same-wave LDS is in-order)
    {
      unsigned short* pp = sP + (size_t)w * 16 * LP;
#pragma unroll
      for (int i = 0; i < 8; ++i) {
        pp[(i + 8 * hi) * LP + r]      = f2bf(s[0][i]);
        pp[(i + 8 * hi) * LP + 16 + r] = f2bf(s[1][i]);
      }
    }
    FragBF pa;
    {
      const unsigned short* ppl = sP + ((size_t)w * 16 + r) * LP + hsel;
      pa.q[0] = *(const u32x4*)ppl;
      pa.q[1] = *(const u32x4*)(ppl + 16);
    }

    // O <- O * alpha + P @ V   (16 feature tiles of 16)
#pragma unroll
    for (int nt = 0; nt < 16; ++nt) {
#pragma unroll
      for (int i = 0; i < 8; ++i) o[nt][i] *= alpha[i];
      FragBF fb;
      const unsigned short* pv = sVt + (feat0 + nt * 16 + r) * LP + hsel;
      fb.q[0] = *(const u32x4*)pv;
      fb.q[1] = *(const u32x4*)(pv + 16);
      o[nt] = __builtin_amdgcn_wmma_f32_16x16x32_bf16(
          false, pa.bf, false, fb.bf, (short)0, o[nt], false, false);
    }
  }

  // epilogue: O / rowsum -> global f32 (residual + LN handled by kernel C)
  float invL[8];
#pragma unroll
  for (int i = 0; i < 8; ++i) invL[i] = 1.f / rowL[i];
#pragma unroll
  for (int nt = 0; nt < 16; ++nt) {
    const int feat = feat0 + nt * 16 + r;
#pragma unroll
    for (int i = 0; i < 8; ++i) {
      const int row = q0 + mg * 16 + i + 8 * hi;
      out[((size_t)b * CDIM + row) * LDIM + feat] = o[nt][i] * invL[i];
    }
  }
}

// ---------------------------------------------------------------------------
// Kernel C: residual (transposed a) + LayerNorm over L, in-place on d_out.
// grid: (C/16, B), block 256; team of 16 lanes per output row.
// ---------------------------------------------------------------------------
__global__ __launch_bounds__(256)
void ln_kernel(const float* __restrict__ a, const float* __restrict__ gamma,
               const float* __restrict__ beta, float* __restrict__ out)
{
  __shared__ float sX[16][LDIM + 1];
  const int b  = blockIdx.y;
  const int c0 = blockIdx.x * 16;
  const int cl = threadIdx.x & 15;
  const int lg = threadIdx.x >> 4;
#pragma unroll 4
  for (int i = 0; i < 32; ++i) {
    const int l = lg + 16 * i;
    sX[cl][l] = a[((size_t)b * LDIM + l) * CDIM + c0 + cl];   // coalesced in c
  }
  __syncthreads();

  const int rrow = threadIdx.x >> 4;
  const int j    = threadIdx.x & 15;
  float* orow = out + ((size_t)b * CDIM + c0 + rrow) * LDIM;

  float xv[32];
  float s1 = 0.f;
#pragma unroll
  for (int i = 0; i < 32; ++i) {
    const int l = j + 16 * i;
    const float x = orow[l] + sX[rrow][l];
    xv[i] = x;
    s1 += x;
  }
  s1 += __shfl_xor(s1, 1, 16);
  s1 += __shfl_xor(s1, 2, 16);
  s1 += __shfl_xor(s1, 4, 16);
  s1 += __shfl_xor(s1, 8, 16);
  const float mu = s1 * (1.f / LDIM);

  float s2 = 0.f;
#pragma unroll
  for (int i = 0; i < 32; ++i) { const float d = xv[i] - mu; s2 += d * d; }
  s2 += __shfl_xor(s2, 1, 16);
  s2 += __shfl_xor(s2, 2, 16);
  s2 += __shfl_xor(s2, 4, 16);
  s2 += __shfl_xor(s2, 8, 16);
  const float inv = rsqrtf(s2 * (1.f / LDIM) + 1e-5f);

#pragma unroll
  for (int i = 0; i < 32; ++i) {
    const int l = j + 16 * i;
    orow[l] = (xv[i] - mu) * inv * gamma[l] + beta[l];
  }
}

// ---------------------------------------------------------------------------
extern "C" void kernel_launch(void* const* d_in, const int* in_sizes, int n_in,
                              void* d_out, int out_size, void* d_ws, size_t ws_size,
                              hipStream_t stream) {
  (void)in_sizes; (void)n_in; (void)out_size; (void)ws_size;
  const float* a     = (const float*)d_in[0];
  const float* bsrc  = (const float*)d_in[1];
  const float* Wq    = (const float*)d_in[2];
  const float* bq    = (const float*)d_in[3];
  const float* Wk    = (const float*)d_in[4];
  const float* bk    = (const float*)d_in[5];
  const float* Wv    = (const float*)d_in[6];
  const float* bv    = (const float*)d_in[7];
  const float* gamma = (const float*)d_in[8];
  const float* beta  = (const float*)d_in[9];
  float* out = (float*)d_out;

  unsigned short* wsQ = (unsigned short*)d_ws;                 // 32 MB each
  unsigned short* wsK = wsQ + (size_t)BATCH * CDIM * LDIM;
  unsigned short* wsV = wsK + (size_t)BATCH * CDIM * LDIM;

  qkv_proj_kernel<<<dim3(CDIM / 64, LDIM / 64, BATCH * 3), 128, 0, stream>>>(
      a, bsrc, Wq, bq, Wk, bk, Wv, bv, wsQ, wsK, wsV);

  const int smemB = (64 * 520 + 32 * 520 + 512 * 40 + 8 * 16 * 40) * 2;  // 151040 B
  hipFuncSetAttribute((const void*)attn_kernel,
                      hipFuncAttributeMaxDynamicSharedMemorySize, smemB);
  attn_kernel<<<dim3(CDIM / 64, BATCH), 256, smemB, stream>>>(wsQ, wsK, wsV, out);

  ln_kernel<<<dim3(CDIM / 16, BATCH), 256, 0, stream>>>(a, gamma, beta, out);
}
